// Net_FG_34256659153245
// MI455X (gfx1250) — matrix operand order repaired and checked by reading, not compiled, not run on hardware
//
#include <hip/hip_runtime.h>

// ---------------------------------------------------------------------------
// MI455X / gfx1250 implementation of the FuncGC graph network.
// Heavy GEMMs (encoder, fem/fam edge MLPs, fnm node MLP) run on the WMMA
// pipe: v_wmma_f32_16x16x32_f16 (f16 A/B, f32 accumulate).
// Edge softmax uses a 2-pass scheme with float atomics (max, then add),
// so no [E,128] scratch is ever materialized.
// Gathers into LDS use gfx1250 GLOBAL_LOAD_ASYNC_TO_LDS_B128 (ASYNCcnt).
// ---------------------------------------------------------------------------

typedef __attribute__((ext_vector_type(16))) _Float16 v16h;
typedef __attribute__((ext_vector_type(8)))  _Float16 v8h;
typedef __attribute__((ext_vector_type(8)))  float    v8f;

#define LEAKY(x) ((x) > 0.0f ? (x) : 0.01f * (x))

// ---- gfx1250 async memory -> LDS (16B per lane), tracked by ASYNCcnt ------
__device__ __forceinline__ void async_ld16(void* lds, const void* gaddr) {
    unsigned off = (unsigned)(uintptr_t)lds;          // addr[31:0] = LDS offset
    unsigned long long ga = (unsigned long long)(uintptr_t)gaddr;
    asm volatile("global_load_async_to_lds_b128 %0, %1, off"
                 :: "v"(off), "v"(ga) : "memory");
}
__device__ __forceinline__ void wait_async0() {
    asm volatile("s_wait_asynccnt 0" ::: "memory");
}

// ---- WMMA fragment loaders (wave32, 16x16x32 f16) -------------------------
// A matrix 16x32 (MxK), row-major source A[16][stride]:
//   lanes 0-15  : row = lane,    K = kb+0..7  and kb+16..23
//   lanes 16-31 : row = lane-16, K = kb+8..15 and kb+24..31
__device__ __forceinline__ v16h fragA_ld(const _Float16* base, int stride,
                                         int kbase, int lane) {
    int r  = lane & 15;
    int hi = lane >> 4;
    const _Float16* p = base + (size_t)r * stride + kbase + hi * 8;
    v8h a0 = *(const v8h*)p;
    v8h a1 = *(const v8h*)(p + 16);
    v16h o;
#pragma unroll
    for (int i = 0; i < 8; ++i) { o[i] = a0[i]; o[i + 8] = a1[i]; }
    return o;
}

// B matrix 32x16 (KxN) from transposed weights Wt[n][K] (K contiguous):
//   lanes 0-15  : col = lane,    K = kb+0..7  and kb+16..23
//   lanes 16-31 : col = lane-16, K = kb+8..15 and kb+24..31
__device__ __forceinline__ v16h fragB_ld(const _Float16* wt, int K,
                                         int kbase, int lane) {
    int c  = lane & 15;
    int hi = lane >> 4;
    const _Float16* p = wt + (size_t)c * K + kbase + hi * 8;
    v8h b0 = *(const v8h*)p;
    v8h b1 = *(const v8h*)(p + 16);
    v16h o;
#pragma unroll
    for (int i = 0; i < 8; ++i) { o[i] = b0[i]; o[i + 8] = b1[i]; }
    return o;
}

// One 16x16 output tile: acc = bias; acc += A(16xK) * Wt(nt-th 16 cols, K)
__device__ __forceinline__ v8f gemm_tile(const _Float16* A, int astride,
                                         const _Float16* Wt, int K,
                                         const float* bias, int nt, int lane) {
    float bv = bias[nt * 16 + (lane & 15)];
    v8f acc;
#pragma unroll
    for (int i = 0; i < 8; ++i) acc[i] = bv;
    const _Float16* B = Wt + (size_t)(nt * 16) * K;
    for (int kb = 0; kb < K; kb += 32) {
        v16h a = fragA_ld(A, astride, kb, lane);
        v16h b = fragB_ld(B, K, kb, lane);
        acc = __builtin_amdgcn_wmma_f32_16x16x32_f16(
            false, a, false, b, (short)0, acc, false, false);
    }
    return acc;
}

// C-fragment element (i) maps to row = i + 8*(lane>>4), col = lane&15.
__device__ __forceinline__ void store_c_f16(v8f acc, _Float16* dst, int stride,
                                            int nt, int lane, bool leaky) {
    int c = lane & 15, hi = lane >> 4;
#pragma unroll
    for (int i = 0; i < 8; ++i) {
        float v = acc[i];
        if (leaky) v = LEAKY(v);
        dst[(size_t)(i + 8 * hi) * stride + nt * 16 + c] = (_Float16)v;
    }
}

// ---- small prep kernels ---------------------------------------------------

// transpose-convert weight f32 [K][Nc] (row-major, as in JAX) -> f16 [Nc][K]
__global__ void k_cvt_wt(const float* __restrict__ src, _Float16* __restrict__ dst,
                         int K, int Nc) {
    int i = blockIdx.x * blockDim.x + threadIdx.x;
    if (i < K * Nc) {
        int k = i / Nc, n = i % Nc;
        dst[(size_t)n * K + k] = (_Float16)src[i];
    }
}

__global__ void k_init(float* mx, float* den, float* num,
                       float* sum_r, float* max_r, int nElems, int gElems) {
    int i = blockIdx.x * blockDim.x + threadIdx.x;
    if (i < nElems) { mx[i] = -INFINITY; den[i] = 0.0f; num[i] = 0.0f; }
    if (i < gElems) { sum_r[i] = 0.0f; max_r[i] = -INFINITY; }
}

// ---- encoder: h = ff(N,64) @ W_enc(64,128) + b ; f16 into hcat[:,0:128] ----
__global__ void k_encoder(const float* __restrict__ ff,
                          const _Float16* __restrict__ enc_wt,  // [128][64]
                          const float* __restrict__ b_enc,
                          _Float16* __restrict__ hcat, int N) {
    __shared__ __align__(16) _Float16 sA[8][16][64];
    int wave = threadIdx.x >> 5, lane = threadIdx.x & 31;
    int tile = blockIdx.x * 8 + wave;
    int base = tile * 16;
    for (int i = lane; i < 16 * 64; i += 32) {
        int r = i >> 6, k = i & 63;
        int row = base + r;
        sA[wave][r][k] = (_Float16)((row < N) ? ff[(size_t)row * 64 + k] : 0.0f);
    }
    __syncthreads();
    int c = lane & 15, hi = lane >> 4;
    bool full = (base + 16 <= N);                // wave-uniform fast path
    for (int nt = 0; nt < 8; ++nt) {
        v8f acc = gemm_tile(&sA[wave][0][0], 64, enc_wt, 64, b_enc, nt, lane);
        _Float16* out = hcat + (size_t)base * 256 + nt * 16 + c;
        if (full) {
#pragma unroll
            for (int i = 0; i < 8; ++i)
                out[(size_t)(i + 8 * hi) * 256] = (_Float16)acc[i];
        } else {
#pragma unroll
            for (int i = 0; i < 8; ++i) {
                int row = base + i + 8 * hi;
                if (row < N)
                    hcat[(size_t)row * 256 + nt * 16 + c] = (_Float16)acc[i];
            }
        }
    }
}

// ---- edge gather: pair rows (h[src] | h[dst]) -> LDS via async DMA --------
__device__ __forceinline__ void gather_pair(const _Float16* hcat,
                                            const int* src, const int* dst,
                                            _Float16 (*sPair)[256], int* sDst,
                                            int ebase, int E, int lane) {
    for (int e = 0; e < 16; ++e) {
        int ge = ebase + e;
        int s = 0, d = 0;
        if (ge < E) { s = src[ge]; d = dst[ge]; }
        if (lane == 0) sDst[e] = d;
        const _Float16* p = (lane < 16)
            ? (hcat + (size_t)s * 256 + lane * 8)
            : (hcat + (size_t)d * 256 + (lane - 16) * 8);
        async_ld16(&sPair[e][lane * 8], p);
    }
    wait_async0();
}

// ---- edge pass A: attention logits -> atomicMax into mx[N,128] ------------
__global__ void __launch_bounds__(64)
k_edge_max(const _Float16* __restrict__ hcat,
           const int* __restrict__ src, const int* __restrict__ dst,
           const _Float16* __restrict__ w1t, const float* __restrict__ b1,
           const _Float16* __restrict__ w2t, const float* __restrict__ b2,
           float* __restrict__ mx, int E) {
    __shared__ __align__(16) _Float16 sPair[2][16][256];
    __shared__ __align__(16) _Float16 sHid[2][16][128];
    __shared__ int sDst[2][16];
    int wave = threadIdx.x >> 5, lane = threadIdx.x & 31;
    int tile = blockIdx.x * 2 + wave;
    int ebase = tile * 16;
    gather_pair(hcat, src, dst, sPair[wave], sDst[wave], ebase, E, lane);
    __syncthreads();
    for (int nt = 0; nt < 8; ++nt) {   // fam layer 1, identity activation
        v8f acc = gemm_tile(&sPair[wave][0][0], 256, w1t, 256, b1, nt, lane);
        store_c_f16(acc, &sHid[wave][0][0], 128, nt, lane, false);
    }
    __syncthreads();
    int c = lane & 15, hi = lane >> 4;
    bool full = (ebase + 16 <= E);               // wave-uniform fast path
    for (int nt = 0; nt < 8; ++nt) {   // fam layer 2 -> logits -> segment max
        v8f acc = gemm_tile(&sHid[wave][0][0], 128, w2t, 128, b2, nt, lane);
        if (full) {
#pragma unroll
            for (int i = 0; i < 8; ++i)
                atomicMax(&mx[(size_t)sDst[wave][i + 8 * hi] * 128 + nt * 16 + c],
                          acc[i]);
        } else {
#pragma unroll
            for (int i = 0; i < 8; ++i) {
                int r = i + 8 * hi;
                if (ebase + r < E)
                    atomicMax(&mx[(size_t)sDst[wave][r] * 128 + nt * 16 + c],
                              acc[i]);
            }
        }
    }
}

// ---- edge pass B: fem messages + fam logits -> den/num atomics ------------
__global__ void __launch_bounds__(64)
k_edge_msg(const _Float16* __restrict__ hcat,
           const int* __restrict__ src, const int* __restrict__ dst,
           const _Float16* __restrict__ fem_w1t, const float* __restrict__ fem_b1,
           const _Float16* __restrict__ fem_w2t, const float* __restrict__ fem_b2,
           const _Float16* __restrict__ fam_w1t, const float* __restrict__ fam_b1,
           const _Float16* __restrict__ fam_w2t, const float* __restrict__ fam_b2,
           const float* __restrict__ mx, float* __restrict__ den,
           float* __restrict__ num, int E) {
    __shared__ __align__(16) _Float16 sPair[2][16][256];
    __shared__ __align__(16) _Float16 sHid[2][16][128];
    __shared__ __align__(16) float    sEmsg[2][16][128];
    __shared__ int sDst[2][16];
    int wave = threadIdx.x >> 5, lane = threadIdx.x & 31;
    int tile = blockIdx.x * 2 + wave;
    int ebase = tile * 16;
    gather_pair(hcat, src, dst, sPair[wave], sDst[wave], ebase, E, lane);
    __syncthreads();
    int c = lane & 15, hi = lane >> 4;
    for (int nt = 0; nt < 8; ++nt) {   // fem layer 1 (leaky)
        v8f acc = gemm_tile(&sPair[wave][0][0], 256, fem_w1t, 256, fem_b1, nt, lane);
        store_c_f16(acc, &sHid[wave][0][0], 128, nt, lane, true);
    }
    __syncthreads();
    for (int nt = 0; nt < 8; ++nt) {   // fem layer 2 -> messages (f32, no act)
        v8f acc = gemm_tile(&sHid[wave][0][0], 128, fem_w2t, 128, fem_b2, nt, lane);
#pragma unroll
        for (int i = 0; i < 8; ++i)
            sEmsg[wave][i + 8 * hi][nt * 16 + c] = acc[i];
    }
    __syncthreads();
    for (int nt = 0; nt < 8; ++nt) {   // fam layer 1 (identity)
        v8f acc = gemm_tile(&sPair[wave][0][0], 256, fam_w1t, 256, fam_b1, nt, lane);
        store_c_f16(acc, &sHid[wave][0][0], 128, nt, lane, false);
    }
    __syncthreads();
    bool full = (ebase + 16 <= E);               // wave-uniform fast path
    for (int nt = 0; nt < 8; ++nt) {   // fam layer 2 -> softmax numerators
        v8f acc = gemm_tile(&sHid[wave][0][0], 128, fam_w2t, 128, fam_b2, nt, lane);
        if (full) {
#pragma unroll
            for (int i = 0; i < 8; ++i) {
                int r = i + 8 * hi;
                size_t off = (size_t)sDst[wave][r] * 128 + nt * 16 + c;
                float ex = __expf(acc[i] - mx[off]);
                atomicAdd(&den[off], ex);
                atomicAdd(&num[off], ex * sEmsg[wave][r][nt * 16 + c]);
            }
        } else {
#pragma unroll
            for (int i = 0; i < 8; ++i) {
                int r = i + 8 * hi;
                if (ebase + r < E) {
                    size_t off = (size_t)sDst[wave][r] * 128 + nt * 16 + c;
                    float ex = __expf(acc[i] - mx[off]);
                    atomicAdd(&den[off], ex);
                    atomicAdd(&num[off], ex * sEmsg[wave][r][nt * 16 + c]);
                }
            }
        }
    }
}

// ---- agg = num/den (guard empty nodes) -> f16 into hcat[:,128:256] --------
__global__ void k_agg(const float* __restrict__ num, const float* __restrict__ den,
                      _Float16* __restrict__ hcat, int N) {
    int i = blockIdx.x * blockDim.x + threadIdx.x;
    if (i < N * 128) {
        int n = i >> 7, j = i & 127;
        float d = den[i];
        float a = (d > 0.0f) ? num[i] / d : 0.0f;
        hcat[(size_t)n * 256 + 128 + j] = (_Float16)a;
    }
}

// ---- fnm node MLP: h2 = (leaky(hcat @ w1 + b1)) @ w2 + b2 -----------------
__global__ void __launch_bounds__(128)
k_fnm(const _Float16* __restrict__ hcat,
      const _Float16* __restrict__ w1t, const float* __restrict__ b1,
      const _Float16* __restrict__ w2t, const float* __restrict__ b2,
      float* __restrict__ h2, int N) {
    __shared__ __align__(16) _Float16 sA[4][16][256];
    __shared__ __align__(16) _Float16 sHid[4][16][128];
    int wave = threadIdx.x >> 5, lane = threadIdx.x & 31;
    int tile = blockIdx.x * 4 + wave;
    int base = tile * 16;
    for (int i = lane; i < 16 * 32; i += 32) {   // stage 16 rows x 256 f16
        int r = i >> 5, ch = i & 31;
        int row = base + r;
        if (row >= N) row = N - 1;               // pad rows (results discarded)
        async_ld16(&sA[wave][r][ch * 8], hcat + (size_t)row * 256 + ch * 8);
    }
    wait_async0();
    __syncthreads();
    for (int nt = 0; nt < 8; ++nt) {             // layer 1 (leaky)
        v8f acc = gemm_tile(&sA[wave][0][0], 256, w1t, 256, b1, nt, lane);
        store_c_f16(acc, &sHid[wave][0][0], 128, nt, lane, true);
    }
    __syncthreads();
    int c = lane & 15, hi = lane >> 4;
    bool full = (base + 16 <= N);                // wave-uniform fast path
    for (int nt = 0; nt < 8; ++nt) {             // layer 2 (no act)
        v8f acc = gemm_tile(&sHid[wave][0][0], 128, w2t, 128, b2, nt, lane);
        if (full) {
            float* out = h2 + (size_t)base * 128 + nt * 16 + c;
#pragma unroll
            for (int i = 0; i < 8; ++i)
                out[(size_t)(i + 8 * hi) * 128] = acc[i];
        } else {
#pragma unroll
            for (int i = 0; i < 8; ++i) {
                int row = base + i + 8 * hi;
                if (row < N) h2[(size_t)row * 128 + nt * 16 + c] = acc[i];
            }
        }
    }
}

// ---- per-graph gated readout: sum(sigmoid(h2@Wg+bg)*h2), max(h2) ----------
__global__ void k_readout(const float* __restrict__ h2,
                          const int* __restrict__ n2g,
                          const float* __restrict__ Wg, const float* __restrict__ bg,
                          float* __restrict__ sum_r, float* __restrict__ max_r,
                          int N) {
    int wave = threadIdx.x >> 5, lane = threadIdx.x & 31;
    int n = blockIdx.x * 8 + wave;
    if (n >= N) return;
    const float* hr = h2 + (size_t)n * 128;
    float v[4];
    float dot = 0.0f;
#pragma unroll
    for (int q = 0; q < 4; ++q) {
        v[q] = hr[lane + 32 * q];
        dot += v[q] * Wg[lane + 32 * q];
    }
    for (int off = 16; off > 0; off >>= 1) dot += __shfl_xor(dot, off, 32);
    float w = 1.0f / (1.0f + __expf(-(dot + bg[0])));
    int g = n2g[n];
    float* sp = sum_r + (size_t)g * 128;
    float* mp = max_r + (size_t)g * 128;
#pragma unroll
    for (int q = 0; q < 4; ++q) {
        atomicAdd(&sp[lane + 32 * q], w * v[q]);
        atomicMax(&mp[lane + 32 * q], v[q]);
    }
}

// ---- regression head: out[g] = leaky([sum,max] @ w1 + b1) @ w2 + b2 -------
__global__ void __launch_bounds__(128)
k_head(const float* __restrict__ sum_r, const float* __restrict__ max_r,
       const float* __restrict__ w1, const float* __restrict__ b1,
       const float* __restrict__ w2, const float* __restrict__ b2,
       float* __restrict__ out, int G) {
    __shared__ float sIn[256];
    __shared__ float sHid[128];
    int g = blockIdx.x;
    int t = threadIdx.x;
    float mv = max_r[(size_t)g * 128 + t];
    if (mv < -1.0e37f) mv = 0.0f;                // empty-graph guard (-inf -> 0)
    sIn[t]       = sum_r[(size_t)g * 128 + t];
    sIn[128 + t] = mv;
    __syncthreads();
    float acc = b1[t];
    for (int k = 0; k < 256; ++k) acc += sIn[k] * w1[k * 128 + t];
    sHid[t] = LEAKY(acc);
    __syncthreads();
    if (t == 0) {
        float o = b2[0];
        for (int j = 0; j < 128; ++j) o += sHid[j] * w2[j];
        out[g] = o;
    }
}

// ---------------------------------------------------------------------------
extern "C" void kernel_launch(void* const* d_in, const int* in_sizes, int n_in,
                              void* d_out, int out_size, void* d_ws, size_t ws_size,
                              hipStream_t stream) {
    (void)n_in; (void)ws_size;
    const float* ff     = (const float*)d_in[0];
    const int*   src    = (const int*)d_in[1];
    const int*   dst    = (const int*)d_in[2];
    const int*   n2g    = (const int*)d_in[3];
    const float* W_enc  = (const float*)d_in[4];
    const float* b_enc  = (const float*)d_in[5];
    const float* fem_w1 = (const float*)d_in[6];
    const float* fem_b1 = (const float*)d_in[7];
    const float* fem_w2 = (const float*)d_in[8];
    const float* fem_b2 = (const float*)d_in[9];
    const float* fnm_w1 = (const float*)d_in[10];
    const float* fnm_b1 = (const float*)d_in[11];
    const float* fnm_w2 = (const float*)d_in[12];
    const float* fnm_b2 = (const float*)d_in[13];
    const float* fam_w1 = (const float*)d_in[14];
    const float* fam_b1 = (const float*)d_in[15];
    const float* fam_w2 = (const float*)d_in[16];
    const float* fam_b2 = (const float*)d_in[17];
    const float* Wg     = (const float*)d_in[18];
    const float* bg     = (const float*)d_in[19];
    const float* reg_w1 = (const float*)d_in[20];
    const float* reg_b1 = (const float*)d_in[21];
    const float* reg_w2 = (const float*)d_in[22];
    const float* reg_b2 = (const float*)d_in[23];

    const int N = in_sizes[0] / 64;   // nodes
    const int E = in_sizes[1];        // edges
    const int G = out_size;           // graphs (OUT = 1)

    // scratch layout (256 B aligned)
    char* w = (char*)d_ws;
    auto alloc = [&](size_t bytes) -> char* {
        char* p = w;
        w += (bytes + 255) & ~(size_t)255;
        return p;
    };
    _Float16* hcat    = (_Float16*)alloc((size_t)N * 256 * 2);  // [h | agg] f16
    float*    mx      = (float*)alloc((size_t)N * 128 * 4);
    float*    den     = (float*)alloc((size_t)N * 128 * 4);
    float*    num     = (float*)alloc((size_t)N * 128 * 4);
    float*    h2      = (float*)alloc((size_t)N * 128 * 4);
    float*    sum_r   = (float*)alloc((size_t)G * 128 * 4);
    float*    max_r   = (float*)alloc((size_t)G * 128 * 4);
    _Float16* enc_wt  = (_Float16*)alloc(128 * 64 * 2);
    _Float16* fem_w1t = (_Float16*)alloc(128 * 256 * 2);
    _Float16* fem_w2t = (_Float16*)alloc(128 * 128 * 2);
    _Float16* fam_w1t = (_Float16*)alloc(128 * 256 * 2);
    _Float16* fam_w2t = (_Float16*)alloc(128 * 128 * 2);
    _Float16* fnm_w1t = (_Float16*)alloc(128 * 256 * 2);
    _Float16* fnm_w2t = (_Float16*)alloc(128 * 128 * 2);

    // weight transpose + f16 conversion (tiny)
    k_cvt_wt<<<(64 * 128 + 255) / 256, 256, 0, stream>>>(W_enc,  enc_wt,  64,  128);
    k_cvt_wt<<<(256 * 128 + 255) / 256, 256, 0, stream>>>(fem_w1, fem_w1t, 256, 128);
    k_cvt_wt<<<(128 * 128 + 255) / 256, 256, 0, stream>>>(fem_w2, fem_w2t, 128, 128);
    k_cvt_wt<<<(256 * 128 + 255) / 256, 256, 0, stream>>>(fam_w1, fam_w1t, 256, 128);
    k_cvt_wt<<<(128 * 128 + 255) / 256, 256, 0, stream>>>(fam_w2, fam_w2t, 128, 128);
    k_cvt_wt<<<(256 * 128 + 255) / 256, 256, 0, stream>>>(fnm_w1, fnm_w1t, 256, 128);
    k_cvt_wt<<<(128 * 128 + 255) / 256, 256, 0, stream>>>(fnm_w2, fnm_w2t, 128, 128);

    // accumulator init (every call: atomics below)
    k_init<<<(N * 128 + 255) / 256, 256, 0, stream>>>(mx, den, num, sum_r, max_r,
                                                      N * 128, G * 128);
    // encoder
    int tilesN = (N + 15) / 16;
    k_encoder<<<(tilesN + 7) / 8, 256, 0, stream>>>(ff, enc_wt, b_enc, hcat, N);

    // edge passes
    int tilesE = (E + 15) / 16;
    int blkE   = (tilesE + 1) / 2;
    k_edge_max<<<blkE, 64, 0, stream>>>(hcat, src, dst, fam_w1t, fam_b1,
                                        fam_w2t, fam_b2, mx, E);
    k_edge_msg<<<blkE, 64, 0, stream>>>(hcat, src, dst,
                                        fem_w1t, fem_b1, fem_w2t, fem_b2,
                                        fam_w1t, fam_b1, fam_w2t, fam_b2,
                                        mx, den, num, E);
    // aggregate
    k_agg<<<(N * 128 + 255) / 256, 256, 0, stream>>>(num, den, hcat, N);
    // node update
    k_fnm<<<(tilesN + 3) / 4, 128, 0, stream>>>(hcat, fnm_w1t, fnm_b1,
                                                fnm_w2t, fnm_b2, h2, N);
    // readout + head
    k_readout<<<(N + 7) / 8, 256, 0, stream>>>(h2, n2g, Wg, bg, sum_r, max_r, N);
    k_head<<<G, 128, 0, stream>>>(sum_r, max_r, reg_w1, reg_b1, reg_w2, reg_b2,
                                  (float*)d_out, G);
}
